// DSSC_15934328668576
// MI455X (gfx1250) — compile-verified, matrix operand order checked by reading, not run on hardware
//
#include <hip/hip_runtime.h>
#include <math.h>

// Problem constants (match reference)
#define NCELLS 30000
#define MPAD   30080            // 235 * 128
#define GENES  2000
#define EDGES_ 480000
#define HEADS  3
#define D0_    256
#define ZDIM   32
#define DDIM   256
#define SIGMA_ 0.1f
#define BN_EPS_ 1e-5f
#define LDT    34               // LDS tile row stride in shorts (68B: 64B data + 4B pad)

typedef __attribute__((ext_vector_type(16))) __bf16        v16bf;
typedef __attribute__((ext_vector_type(8)))  float         v8f;
typedef __attribute__((ext_vector_type(8)))  unsigned int  v8u;

#if defined(__HIP_DEVICE_COMPILE__) && __has_builtin(__builtin_amdgcn_tensor_load_to_lds)
#define HAVE_TDM 1
#else
#define HAVE_TDM 0
#endif

__device__ __forceinline__ unsigned short f2bf(float f) {
  unsigned int u = __float_as_uint(f);
  unsigned int r = (u + 0x7FFFu + ((u >> 16) & 1u)) >> 16;   // RNE
  return (unsigned short)r;
}

__device__ __forceinline__ void atomicMaxF(float* addr, float val) {
  unsigned int* ua = (unsigned int*)addr;
  unsigned int old = *ua;
  while (__uint_as_float(old) < val) {
    unsigned int assumed = old;
    old = atomicCAS(ua, assumed, __float_as_uint(val));
    if (old == assumed) break;
  }
}

// ---------------------------------------------------------------- fill
__global__ void k_fill(float* p, float v, long long n) {
  long long i = blockIdx.x * (long long)blockDim.x + threadIdx.x;
  long long s = (long long)gridDim.x * blockDim.x;
  for (; i < n; i += s) p[i] = v;
}

// ----------------------------------------------------- f32 -> padded bf16
// dst[Mpad][Kpad] <- src[M][K] (+ sigma*noise), zero-padded
__global__ void k_cvt_A(const float* __restrict__ A, const float* __restrict__ noise,
                        unsigned short* __restrict__ dst, int M, int K, int Kpad,
                        long long total)
{
  long long i = blockIdx.x * (long long)blockDim.x + threadIdx.x;
  long long s = (long long)gridDim.x * blockDim.x;
  for (; i < total; i += s) {
    int r = (int)(i / Kpad), c = (int)(i % Kpad);
    float v = 0.f;
    if (r < M && c < K) {
      long long gi = (long long)r * K + c;
      v = A[gi];
      if (noise) v += SIGMA_ * noise[gi];
    }
    dst[i] = f2bf(v);
  }
}

// dst[Ncpad][Kpad] <- transpose(W[K][Nc]) as bf16, zero-padded
__global__ void k_cvt_WT(const float* __restrict__ W, unsigned short* __restrict__ dst,
                         int K, int Nc, int Kpad, long long total)
{
  long long i = blockIdx.x * (long long)blockDim.x + threadIdx.x;
  long long s = (long long)gridDim.x * blockDim.x;
  for (; i < total; i += s) {
    int n = (int)(i / Kpad), k = (int)(i % Kpad);
    float v = (n < Nc && k < K) ? W[(long long)k * Nc + n] : 0.f;
    dst[i] = f2bf(v);
  }
}

// -------------------------------------------------- TDM tile load (D# build)
#if HAVE_TDM
typedef __attribute__((ext_vector_type(4))) unsigned int tdm_u32x4;
typedef __attribute__((ext_vector_type(8))) int          tdm_i32x8;
typedef __attribute__((ext_vector_type(4))) int          tdm_i32x4;

__device__ __forceinline__ void tdm_load_tile_bf16(
    unsigned lds_byte_addr, const unsigned short* gptr,
    unsigned tileK, unsigned tileRows, unsigned strideElems)
{
  unsigned long long ga = (unsigned long long)(size_t)gptr;
  tdm_u32x4 g0;
  g0.x = 1u;                                            // count=1, user-mode
  g0.y = lds_byte_addr;                                 // LDS dest (bytes)
  g0.z = (unsigned)ga;                                  // global_addr[31:0]
  g0.w = (unsigned)((ga >> 32) & 0x01FFFFFFu) | (2u << 30);  // addr[56:32] | type=2
  const unsigned td0 = strideElems;                     // padded -> never OOB
  const unsigned td1 = 0x7FFFFFFFu;
  tdm_i32x8 g1;
  g1[0] = (int)0x00D10000u;     // data_size=2B | pad_enable | pad_interval=16DW (pad=1DW)
  g1[1] = (int)((td0 & 0xFFFFu) << 16);                 // tensor_dim0[15:0]
  g1[2] = (int)((td0 >> 16) | ((td1 & 0xFFFFu) << 16)); // dim0 hi | dim1 lo
  g1[3] = (int)((td1 >> 16) | (tileK << 16));           // dim1 hi | tile_dim0
  g1[4] = (int)(tileRows & 0xFFFFu);                    // tile_dim1 (tile_dim2=0)
  g1[5] = (int)strideElems;                             // tensor_dim0_stride[31:0]
  g1[6] = 0;
  g1[7] = 0;
  tdm_i32x4 zz = {0, 0, 0, 0};
  tdm_i32x8 z8 = {0, 0, 0, 0, 0, 0, 0, 0};
  // amdgpu-toolchain (clang-23) 6-arg form: (g0, g1, g2, g3, extra, cpol)
  __builtin_amdgcn_tensor_load_to_lds(g0, g1, zz, zz, z8, 0);
}
#endif

__device__ __forceinline__ void stage_tile(unsigned short* smem, unsigned ldsBase,
    int eA, int eB, const unsigned short* Aptr, const unsigned short* Bptr,
    int Kpad, int tid)
{
#if HAVE_TDM
  (void)smem;
  if (tid == 0) {
    tdm_load_tile_bf16(ldsBase + 2u * (unsigned)eA, Aptr, 32u, 128u, (unsigned)Kpad);
    tdm_load_tile_bf16(ldsBase + 2u * (unsigned)eB, Bptr, 32u, 64u,  (unsigned)Kpad);
  }
#else
  (void)ldsBase;
  for (int i = tid; i < 128 * 32; i += 256) {
    int r = i >> 5, c = i & 31;
    smem[eA + r * LDT + c] = Aptr[(long long)r * Kpad + c];
  }
  for (int i = tid; i < 64 * 32; i += 256) {
    int r = i >> 5, c = i & 31;
    smem[eB + r * LDT + c] = Bptr[(long long)r * Kpad + c];
  }
#endif
}

// ------------------------------------------------- WMMA bf16 GEMM, TDM-staged
// C[M,Nc](f32) = epi( Abf[Mpad,Kpad] @ Bbf[Ncpad,Kpad]^T + bias )
// Block 256 thr = 8 waves; block tile 128(M) x 64(N); wave tile 16 x 64.
// Double-buffered LDS; A/B tiles DMA'd by the Tensor Data Mover.
template <int EPI>
__global__ __launch_bounds__(256)
void k_gemm2(const unsigned short* __restrict__ Abf, const unsigned short* __restrict__ Bbf,
             const float* __restrict__ bias, float* __restrict__ C,
             int M, int Nc, int Kpad)
{
  __shared__ unsigned short smem[2 * 128 * LDT + 2 * 64 * LDT];
  const int eA0 = 0, eA1 = 128 * LDT;
  const int eB0 = 2 * 128 * LDT, eB1 = 2 * 128 * LDT + 64 * LDT;
  const int eA[2] = {eA0, eA1};
  const int eB[2] = {eB0, eB1};
  unsigned ldsBase = (unsigned)(size_t)&smem[0];

  const int tid  = threadIdx.x;
  const int lane = tid & 31;
  const int w    = tid >> 5;
  const int l15  = lane & 15;
  const int half = lane >> 4;
  const int m0   = blockIdx.x * 128;
  const int n0   = blockIdx.y * 64;

  const unsigned short* Abase = Abf + (long long)m0 * Kpad;
  const unsigned short* Bbase = Bbf + (long long)n0 * Kpad;

  v8f acc[4];
  const v8f vzero = {0.f, 0.f, 0.f, 0.f, 0.f, 0.f, 0.f, 0.f};
#pragma unroll
  for (int t = 0; t < 4; ++t) acc[t] = vzero;

  const int ksteps = Kpad >> 5;
  stage_tile(smem, ldsBase, eA[0], eB[0], Abase, Bbase, Kpad, tid);

  for (int ks = 0; ks < ksteps; ++ks) {
    const int buf = ks & 1;
    if (ks + 1 < ksteps) {
      stage_tile(smem, ldsBase, eA[(ks + 1) & 1], eB[(ks + 1) & 1],
                 Abase + (ks + 1) * 32, Bbase + (ks + 1) * 32, Kpad, tid);
#if HAVE_TDM
      __builtin_amdgcn_s_wait_tensorcnt((short)2);   // tile ks landed; ks+1 in flight
#endif
    } else {
#if HAVE_TDM
      __builtin_amdgcn_s_wait_tensorcnt((short)0);
#endif
    }
    __syncthreads();

    // A fragment: lane l15 = M row, half selects K-halves (ISA 16-bit A layout)
    v8u au;
    const int arow = eA[buf] + (w * 16 + l15) * LDT;
#pragma unroll
    for (int j = 0; j < 8; ++j) {
      int kb = ((j & 4) ? 16 : 0) + half * 8 + ((j & 3) << 1);
      au[j] = *(const unsigned int*)(&smem[arow + kb]);
    }
    v16bf afrag = __builtin_bit_cast(v16bf, au);

#pragma unroll
    for (int t = 0; t < 4; ++t) {
      v8u bu;
      const int brow = eB[buf] + (t * 16 + l15) * LDT;
#pragma unroll
      for (int j = 0; j < 8; ++j) {
        int kb = ((j & 4) ? 16 : 0) + half * 8 + ((j & 3) << 1);
        bu[j] = *(const unsigned int*)(&smem[brow + kb]);
      }
      v16bf bfrag = __builtin_bit_cast(v16bf, bu);
      acc[t] = __builtin_amdgcn_wmma_f32_16x16x32_bf16(
                 false, afrag, false, bfrag, (short)0, acc[t], false, false);
    }
    __syncthreads();
  }

  // epilogue: C/D layout -> VGPR r: M = r + 8*half ; N = l15
#pragma unroll
  for (int t = 0; t < 4; ++t) {
    int col = n0 + t * 16 + l15;
    if (col >= Nc) continue;
    float bv = bias ? bias[col] : 0.0f;
#pragma unroll
    for (int r = 0; r < 8; ++r) {
      int row = m0 + w * 16 + r + 8 * half;
      if (row >= M) continue;
      float v = acc[t][r] + bv;
      if (EPI == 1)      { v = expf(v); v = fminf(fmaxf(v, 1e-5f), 1e6f); }
      else if (EPI == 2) { v = (v > 20.f) ? v : log1pf(expf(v));
                           v = fminf(fmaxf(v, 1e-4f), 1e4f); }
      else if (EPI == 3) { v = 1.0f / (1.0f + expf(-v)); }
      C[(long long)row * Nc + col] = v;
    }
  }
}

// --------------------------------------------------------- attention scores
__global__ void k_attn(const float* __restrict__ h, const float* __restrict__ al,
                       const float* __restrict__ ar, float* __restrict__ el,
                       float* __restrict__ er, int n, int H, int D)
{
  int i = blockIdx.x * blockDim.x + threadIdx.x;
  if (i >= n * H) return;
  int node = i / H, hh = i % H;
  const float* row = h + (long long)node * H * D + hh * D;
  const float* a = al + hh * D;
  const float* b = ar + hh * D;
  float sl = 0.f, sr = 0.f;
  for (int d = 0; d < D; ++d) { float v = row[d]; sl += v * a[d]; sr += v * b[d]; }
  el[i] = sl; er[i] = sr;
}

// ------------------------------------------------------------- edge softmax
__global__ void k_edge_scores(const float* __restrict__ el, const float* __restrict__ er,
                              const int* __restrict__ src, const int* __restrict__ dst,
                              float* __restrict__ sc, float* __restrict__ mbuf, int E, int H)
{
  int e = blockIdx.x * blockDim.x + threadIdx.x;
  if (e >= E) return;
  int s = src[e], d = dst[e];
  for (int h = 0; h < H; ++h) {
    float v = el[s * H + h] + er[d * H + h];
    v = (v > 0.f) ? v : 0.2f * v;                       // leaky_relu 0.2
    sc[(long long)e * H + h] = v;
    atomicMaxF(&mbuf[d * H + h], v);
  }
}

__global__ void k_fixm(float* m, int n) {               // zero-in-degree safe
  int i = blockIdx.x * blockDim.x + threadIdx.x;
  if (i < n && !(m[i] > -1e38f)) m[i] = 0.0f;
}

__global__ void k_edge_expsum(float* __restrict__ sc, const float* __restrict__ mbuf,
                              const int* __restrict__ dst, float* __restrict__ ssum,
                              int E, int H)
{
  int e = blockIdx.x * blockDim.x + threadIdx.x;
  if (e >= E) return;
  int d = dst[e];
  for (int h = 0; h < H; ++h) {
    float ex = expf(sc[(long long)e * H + h] - mbuf[d * H + h]);
    sc[(long long)e * H + h] = ex;
    atomicAdd(&ssum[d * H + h], ex);
  }
}

// one block per edge; threads sweep H*D features
__global__ void k_edge_agg(const float* __restrict__ hproj, const float* __restrict__ ex,
                           const float* __restrict__ ssum, const int* __restrict__ src,
                           const int* __restrict__ dst, float* __restrict__ out,
                           int E, int H, int D)
{
  int e = blockIdx.x;
  int s = src[e], d = dst[e];
  int HD = H * D;
  for (int idx = threadIdx.x; idx < HD; idx += blockDim.x) {
    int h = idx / D;
    float w = ex[(long long)e * H + h] / ssum[d * H + h];
    atomicAdd(&out[(long long)d * HD + idx], w * hproj[(long long)s * HD + idx]);
  }
}

// bias + (optional elu) per (n,h,d), then sum over heads
__global__ void k_head_reduce(const float* __restrict__ out, const float* __restrict__ bias,
                              float* __restrict__ res, int n, int H, int D, int do_elu)
{
  long long i = blockIdx.x * (long long)blockDim.x + threadIdx.x;
  long long tot = (long long)n * D;
  if (i >= tot) return;
  int node = (int)(i / D), d = (int)(i % D);
  float s = 0.f;
  for (int h = 0; h < H; ++h) {
    float v = out[(long long)node * H * D + h * D + d] + bias[h * D + d];
    if (do_elu) v = (v > 0.f) ? v : (expf(v) - 1.0f);
    s += v;
  }
  res[(long long)node * D + d] = s;
}

// --------------------------------------------------------------- batchnorm
__global__ void k_bn_stats(const float* __restrict__ x, int n, int C,
                           float* __restrict__ mean, float* __restrict__ var)
{
  __shared__ float s1[256], s2[256];
  int c = blockIdx.x, tid = threadIdx.x;
  float a = 0.f, b = 0.f;
  for (int r = tid; r < n; r += blockDim.x) {
    float v = x[(long long)r * C + c];
    a += v; b += v * v;
  }
  s1[tid] = a; s2[tid] = b; __syncthreads();
  for (int off = 128; off > 0; off >>= 1) {
    if (tid < off) { s1[tid] += s1[tid + off]; s2[tid] += s2[tid + off]; }
    __syncthreads();
  }
  if (tid == 0) {
    float mu = s1[0] / n;
    mean[c] = mu;
    var[c]  = s2[0] / n - mu * mu;   // biased, matches reference
  }
}

__global__ void k_bn_apply(float* __restrict__ x, const float* __restrict__ mean,
                           const float* __restrict__ var, const float* __restrict__ g,
                           const float* __restrict__ b, long long total, int C, int do_elu)
{
  long long i = blockIdx.x * (long long)blockDim.x + threadIdx.x;
  if (i >= total) return;
  int c = (int)(i % C);
  float v = (x[i] - mean[c]) * rsqrtf(var[c] + BN_EPS_) * g[c] + b[c];
  if (do_elu) v = (v > 0.f) ? v : (expf(v) - 1.0f);
  x[i] = v;
}

// =============================================================== host side
namespace {

struct Ctx {
  hipStream_t st;
  const int *src, *dst;
  // f32 workspace
  float *hproj, *outacc, *el, *er, *m, *ss, *ex, *hsum, *z, *t, *mean, *var;
  // bf16 workspace
  unsigned short *Abf;                                   // [MPAD x Kpad] staging
  unsigned short *g0WT, *gmWT, *decWT, *meanWT, *dispWT, *piWT;
};

inline int idiv(long long a, int b) { return (int)((a + b - 1) / b); }
inline int kpad(int k)  { return (k + 31) & ~31; }
inline int ncpad(int n) { return (n + 63) & ~63; }

void cvt_A(const Ctx& c, const float* A, const float* noise, int M, int K) {
  int Kp = kpad(K);
  long long total = (long long)MPAD * Kp;
  k_cvt_A<<<8192, 256, 0, c.st>>>(A, noise, c.Abf, M, K, Kp, total);
}

void cvt_WT(const Ctx& c, const float* W, unsigned short* dst, int K, int Nc) {
  int Kp = kpad(K), Np = ncpad(Nc);
  long long total = (long long)Np * Kp;
  k_cvt_WT<<<4096, 256, 0, c.st>>>(W, dst, K, Nc, Kp, total);
}

void launch_gemm(const Ctx& c, int epi, const unsigned short* Bbf, const float* bias,
                 float* C, int M, int Nc, int K)
{
  int Kp = kpad(K), Np = ncpad(Nc);
  dim3 g(MPAD / 128, Np / 64), b(256);
  switch (epi) {
    case 0: k_gemm2<0><<<g, b, 0, c.st>>>(c.Abf, Bbf, bias, C, M, Nc, Kp); break;
    case 1: k_gemm2<1><<<g, b, 0, c.st>>>(c.Abf, Bbf, bias, C, M, Nc, Kp); break;
    case 2: k_gemm2<2><<<g, b, 0, c.st>>>(c.Abf, Bbf, bias, C, M, Nc, Kp); break;
    case 3: k_gemm2<3><<<g, b, 0, c.st>>>(c.Abf, Bbf, bias, C, M, Nc, Kp); break;
  }
}

void gat_layer(const Ctx& c, const float* proj /* [N,H*D] */, const float* al,
               const float* ar, const float* bias, float* res /* [N,D] */,
               int D, int do_elu)
{
  const int NH = NCELLS * HEADS;
  k_attn<<<idiv(NH, 256), 256, 0, c.st>>>(proj, al, ar, c.el, c.er, NCELLS, HEADS, D);
  k_fill<<<1024, 256, 0, c.st>>>(c.m, -INFINITY, NH);
  k_fill<<<1024, 256, 0, c.st>>>(c.ss, 0.0f, NH);
  k_fill<<<8192, 256, 0, c.st>>>(c.outacc, 0.0f, (long long)NCELLS * HEADS * D);
  k_edge_scores<<<idiv(EDGES_, 256), 256, 0, c.st>>>(c.el, c.er, c.src, c.dst,
                                                     c.ex, c.m, EDGES_, HEADS);
  k_fixm<<<idiv(NH, 256), 256, 0, c.st>>>(c.m, NH);
  k_edge_expsum<<<idiv(EDGES_, 256), 256, 0, c.st>>>(c.ex, c.m, c.dst, c.ss, EDGES_, HEADS);
  int aggThreads = (HEADS * D >= 256) ? 256 : HEADS * D;
  k_edge_agg<<<EDGES_, aggThreads, 0, c.st>>>(proj, c.ex, c.ss, c.src, c.dst,
                                              c.outacc, EDGES_, HEADS, D);
  k_head_reduce<<<idiv((long long)NCELLS * D, 256), 256, 0, c.st>>>(
      c.outacc, bias, res, NCELLS, HEADS, D, do_elu);
}

void bn_inplace(const Ctx& c, float* x, const float* g, const float* b, int C, int do_elu)
{
  k_bn_stats<<<C, 256, 0, c.st>>>(x, NCELLS, C, c.mean, c.var);
  k_bn_apply<<<idiv((long long)NCELLS * C, 256), 256, 0, c.st>>>(
      x, c.mean, c.var, g, b, (long long)NCELLS * C, C, do_elu);
}

void encoder(const Ctx& c, const float* x, const float* noise,
             const float* al0, const float* ar0, const float* b0,
             const float* bng, const float* bnb,
             const float* almu, const float* armu, const float* bmu,
             float* zout)
{
  // GAT layer 0: [N,G] @ [G, H*D0]
  cvt_A(c, x, noise, NCELLS, GENES);
  launch_gemm(c, 0, c.g0WT, nullptr, c.hproj, NCELLS, HEADS * D0_, GENES);
  gat_layer(c, c.hproj, al0, ar0, b0, c.hsum, D0_, /*elu=*/1);
  // double BN (faithful to source bug)
  bn_inplace(c, c.hsum, bng, bnb, D0_, 0);
  bn_inplace(c, c.hsum, bng, bnb, D0_, 0);
  // GAT mu: [N,D0] @ [D0, H*Z]
  cvt_A(c, c.hsum, nullptr, NCELLS, D0_);
  launch_gemm(c, 0, c.gmWT, nullptr, c.hproj, NCELLS, HEADS * ZDIM, D0_);
  gat_layer(c, c.hproj, almu, armu, bmu, zout, ZDIM, /*elu=*/0);
}

} // namespace

extern "C" void kernel_launch(void* const* d_in, const int* in_sizes, int n_in,
                              void* d_out, int out_size, void* d_ws, size_t ws_size,
                              hipStream_t stream)
{
  (void)in_sizes; (void)n_in; (void)out_size; (void)ws_size;
  // inputs in setup_inputs() insertion order
  const float* x     = (const float*)d_in[0];
  const float* noise = (const float*)d_in[1];
  const int*   src   = (const int*)d_in[2];
  const int*   dst   = (const int*)d_in[3];
  const float* g0W   = (const float*)d_in[4];
  const float* g0al  = (const float*)d_in[5];
  const float* g0ar  = (const float*)d_in[6];
  const float* g0b   = (const float*)d_in[7];
  const float* bn0g  = (const float*)d_in[8];
  const float* bn0b  = (const float*)d_in[9];
  const float* gmW   = (const float*)d_in[10];
  const float* gmal  = (const float*)d_in[11];
  const float* gmar  = (const float*)d_in[12];
  const float* gmb   = (const float*)d_in[13];
  const float* decW  = (const float*)d_in[14];
  const float* decb  = (const float*)d_in[15];
  const float* bngW  = (const float*)d_in[16];
  const float* bnbW  = (const float*)d_in[17];
  const float* meanW = (const float*)d_in[18];
  const float* meanb = (const float*)d_in[19];
  const float* dispW = (const float*)d_in[20];
  const float* dispb = (const float*)d_in[21];
  const float* piW   = (const float*)d_in[22];
  const float* pib   = (const float*)d_in[23];

  float* out = (float*)d_out;
  const long long OFF_MEAN = (long long)NCELLS * ZDIM;
  const long long OFF_DISP = OFF_MEAN + (long long)NCELLS * GENES;
  const long long OFF_PI   = OFF_DISP + (long long)NCELLS * GENES;

  // workspace layout (float units, 64-float aligned blocks)
  float* w = (float*)d_ws;
  Ctx c;
  c.st = stream; c.src = src; c.dst = dst;
  long long off = 0;
  auto take = [&](long long nfloats) {
    float* p = w + off;
    off += (nfloats + 63) & ~63LL;
    return p;
  };
  c.hproj  = take((long long)NCELLS * HEADS * D0_);
  c.outacc = take((long long)NCELLS * HEADS * D0_);
  c.el     = take((long long)NCELLS * HEADS);
  c.er     = take((long long)NCELLS * HEADS);
  c.m      = take((long long)NCELLS * HEADS);
  c.ss     = take((long long)NCELLS * HEADS);
  c.ex     = take((long long)EDGES_ * HEADS);
  c.hsum   = take((long long)NCELLS * D0_);
  c.z      = take((long long)NCELLS * ZDIM);
  c.t      = take((long long)NCELLS * DDIM);
  c.mean   = take(512);
  c.var    = take(512);
  c.Abf    = (unsigned short*)take((long long)MPAD * 2016 / 2);       // max Kpad = 2016
  c.g0WT   = (unsigned short*)take((long long)768  * 2016 / 2);
  c.gmWT   = (unsigned short*)take((long long)128  * 256  / 2);
  c.decWT  = (unsigned short*)take((long long)256  * 32   / 2);
  c.meanWT = (unsigned short*)take((long long)2048 * 256  / 2);
  c.dispWT = (unsigned short*)take((long long)2048 * 256  / 2);
  c.piWT   = (unsigned short*)take((long long)2048 * 256  / 2);

  // ---- weights -> transposed bf16 (once per call) ----
  cvt_WT(c, g0W,   c.g0WT,   GENES, HEADS * D0_);
  cvt_WT(c, gmW,   c.gmWT,   D0_,   HEADS * ZDIM);
  cvt_WT(c, decW,  c.decWT,  ZDIM,  DDIM);
  cvt_WT(c, meanW, c.meanWT, DDIM,  GENES);
  cvt_WT(c, dispW, c.dispWT, DDIM,  GENES);
  cvt_WT(c, piW,   c.piWT,   DDIM,  GENES);

  // ---- noisy pass: z = encoder(x + sigma*noise) ----
  encoder(c, x, noise, g0al, g0ar, g0b, bn0g, bn0b, gmal, gmar, gmb, c.z);

  // ---- decoder ----
  cvt_A(c, c.z, nullptr, NCELLS, ZDIM);
  launch_gemm(c, 0, c.decWT, decb, c.t, NCELLS, DDIM, ZDIM);
  bn_inplace(c, c.t, bngW, bnbW, DDIM, /*elu=*/1);
  cvt_A(c, c.t, nullptr, NCELLS, DDIM);
  launch_gemm(c, 1, c.meanWT, meanb, out + OFF_MEAN, NCELLS, GENES, DDIM);
  launch_gemm(c, 2, c.dispWT, dispb, out + OFF_DISP, NCELLS, GENES, DDIM);
  launch_gemm(c, 3, c.piWT,   pib,   out + OFF_PI,   NCELLS, GENES, DDIM);

  // ---- clean pass: z0 = encoder(x) -> d_out[0:N*Z) ----
  encoder(c, x, nullptr, g0al, g0ar, g0b, bn0g, bn0b, gmal, gmar, gmb, out);
}
